// SpatioTemporalGNN_76982993813634
// MI455X (gfx1250) — compile-verified, matrix operand order checked by reading, not run on hardware
//
#include <hip/hip_runtime.h>
#include <hip/hip_bf16.h>

#define DFEAT 256

typedef __attribute__((ext_vector_type(16))) __bf16 v16bf;
typedef __attribute__((ext_vector_type(8)))  float  v8f;

// round-to-nearest-even fp32 -> bf16 (bit trick)
__device__ __forceinline__ unsigned short f2bf(float f) {
    unsigned int u = __float_as_uint(f);
    u += 0x7FFFu + ((u >> 16) & 1u);
    return (unsigned short)(u >> 16);
}

__device__ __forceinline__ unsigned int pk2bf(float f0, float f1) {
    return (unsigned int)f2bf(f0) | ((unsigned int)f2bf(f1) << 16);
}

__global__ void zero_f32(float* __restrict__ p, int n) {
    int i = blockIdx.x * blockDim.x + threadIdx.x;
    int stride = gridDim.x * blockDim.x;
    for (; i < n; i += stride) p[i] = 0.0f;
}

__global__ void degree_kernel(const int* __restrict__ dst, float* __restrict__ deg, int ne) {
    int i = blockIdx.x * blockDim.x + threadIdx.x;
    int stride = gridDim.x * blockDim.x;
    for (; i < ne; i += stride) unsafeAtomicAdd(&deg[dst[i]], 1.0f);
}

__global__ void invdeg_kernel(const float* __restrict__ deg, float* __restrict__ inv, int n) {
    int i = blockIdx.x * blockDim.x + threadIdx.x;
    int stride = gridDim.x * blockDim.x;
    for (; i < n; i += stride) inv[i] = 1.0f / fmaxf(deg[i], 1.0f);
}

// one wave (32 lanes) per edge; each lane moves 8 floats (2x float4) with fp32 atomics
__global__ __launch_bounds__(256) void scatter_kernel(const int* __restrict__ src,
                                                      const int* __restrict__ dst,
                                                      const float* __restrict__ x,
                                                      float* __restrict__ agg, int ne) {
    int e = blockIdx.x * 8 + (threadIdx.x >> 5);
    int lane = threadIdx.x & 31;
    if (e >= ne) return;
    int s = src[e], d = dst[e];
    const float4* xs = (const float4*)(x + (size_t)s * DFEAT);
    float* ad = agg + (size_t)d * DFEAT;
#pragma unroll
    for (int j = 0; j < 2; ++j) {
        float4 v = xs[lane + 32 * j];
        int base = (lane + 32 * j) * 4;
        unsafeAtomicAdd(ad + base + 0, v.x);
        unsafeAtomicAdd(ad + base + 1, v.y);
        unsafeAtomicAdd(ad + base + 2, v.z);
        unsafeAtomicAdd(ad + base + 3, v.w);
    }
}

// Wb[0:65536) = bf16(Wl) row-major [n][k], Wb[65536:131072) = bf16(Wr) row-major
__global__ void wconv(const float* __restrict__ Wl, const float* __restrict__ Wr,
                      unsigned short* __restrict__ Wb) {
    int i = blockIdx.x * blockDim.x + threadIdx.x;  // 65536 threads
    Wb[i] = f2bf(Wl[i]);
    Wb[65536 + i] = f2bf(Wr[i]);
}

// Fused SAGE layer GEMM:
//   out[i,n] = relu?( sum_k (agg[i,k]*inv[i]) * Wl[n,k] + sum_k x[i,k] * Wr[n,k] + bias[n] )
// A operands are converted fp32 -> bf16 in-register (mean scaling fused), no staging buffer.
// Block = 8 waves; each wave owns 16 rows x all 256 cols (16 16x16 col-tiles resident,
// 128 accumulator VGPRs), so each A fragment feeds 16 v_wmma issues.
__global__ __launch_bounds__(256) void gemm_kernel(const float* __restrict__ agg,
                                                   const float* __restrict__ inv,
                                                   const float* __restrict__ xin,
                                                   const unsigned short* __restrict__ Wb,
                                                   const float* __restrict__ bias,
                                                   float* __restrict__ out, int nrows, int relu) {
    const int wave = threadIdx.x >> 5;
    const int lane = threadIdx.x & 31;
    const int half = lane >> 4;   // 0: lanes 0-15, 1: lanes 16-31
    const int l16  = lane & 15;
    const int rowBase = blockIdx.x * 128 + wave * 16;

    int arow = rowBase + l16;
    if (arow >= nrows) arow = nrows - 1;  // clamp loads; stores are guarded
    const float iv = inv[arow];

    union Frag { uint4 u[2]; unsigned int w[8]; v16bf v; };

    v8f acc[16];
#pragma unroll
    for (int nt = 0; nt < 16; ++nt)
#pragma unroll
        for (int r = 0; r < 8; ++r) acc[nt][r] = 0.0f;

    for (int h = 0; h < 2; ++h) {
        const float* ap = (h ? xin : agg) + (size_t)arow * DFEAT;
        const float sc = h ? 1.0f : iv;
        const unsigned short* wh = Wb + h * 65536;  // h=0: Wl, h=1: Wr

        for (int k0 = 0; k0 < 256; k0 += 32) {
            // A fragment: lanes 0-15 hold K {k0..k0+7, k0+16..k0+23}; lanes 16-31 the rest
            const float* c0 = ap + k0 + half * 8;
            const float* c1 = ap + k0 + 16 + half * 8;
            float4 a0 = *(const float4*)(c0);
            float4 a1 = *(const float4*)(c0 + 4);
            float4 a2 = *(const float4*)(c1);
            float4 a3 = *(const float4*)(c1 + 4);
            Frag af;
            af.w[0] = pk2bf(a0.x * sc, a0.y * sc);
            af.w[1] = pk2bf(a0.z * sc, a0.w * sc);
            af.w[2] = pk2bf(a1.x * sc, a1.y * sc);
            af.w[3] = pk2bf(a1.z * sc, a1.w * sc);
            af.w[4] = pk2bf(a2.x * sc, a2.y * sc);
            af.w[5] = pk2bf(a2.z * sc, a2.w * sc);
            af.w[6] = pk2bf(a3.x * sc, a3.y * sc);
            af.w[7] = pk2bf(a3.z * sc, a3.w * sc);
#pragma unroll
            for (int nt = 0; nt < 16; ++nt) {
                const unsigned short* bp = wh + (size_t)(nt * 16 + l16) * 256 + k0;
                Frag bf;
                bf.u[0] = *(const uint4*)(bp + half * 8);
                bf.u[1] = *(const uint4*)(bp + 16 + half * 8);
                acc[nt] = __builtin_amdgcn_wmma_f32_16x16x32_bf16(
                    false, af.v, false, bf.v, (short)0, acc[nt], false, false);
            }
        }
    }

#pragma unroll
    for (int nt = 0; nt < 16; ++nt) {
        int col = nt * 16 + l16;
        float bc = bias[col];
#pragma unroll
        for (int r = 0; r < 8; ++r) {
            int row = rowBase + half * 8 + r;  // D-tile: vgpr r -> M = r + half*8
            float v = acc[nt][r] + bc;
            if (relu) v = fmaxf(v, 0.0f);
            if (row < nrows) out[(size_t)row * DFEAT + col] = v;
        }
    }
}

extern "C" void kernel_launch(void* const* d_in, const int* in_sizes, int n_in,
                              void* d_out, int out_size, void* d_ws, size_t ws_size,
                              hipStream_t stream) {
    const int* ei = (const int*)d_in[0];
    const float* x_all = (const float*)d_in[1];
    const float* Wl[3] = {(const float*)d_in[2], (const float*)d_in[5], (const float*)d_in[8]};
    const float* Wr[3] = {(const float*)d_in[3], (const float*)d_in[6], (const float*)d_in[9]};
    const float* bb[3] = {(const float*)d_in[4], (const float*)d_in[7], (const float*)d_in[10]};
    const int E = in_sizes[0] / 2;
    const int N = in_sizes[1] / DFEAT;
    const int* src = ei;
    const int* dst = ei + E;

    char* ws = (char*)d_ws;
    size_t off = 0;
    auto alloc = [&](size_t bytes) -> void* {
        void* p = ws + off;
        off = (off + bytes + 255) & ~(size_t)255;
        return p;
    };

    float* agg = (float*)alloc((size_t)N * DFEAT * sizeof(float));
    float* xb0 = (float*)alloc((size_t)N * DFEAT * sizeof(float));
    float* xb1 = (float*)alloc((size_t)N * DFEAT * sizeof(float));
    float* deg = (float*)alloc((size_t)N * sizeof(float));
    float* inv = (float*)alloc((size_t)N * sizeof(float));
    unsigned short* Wb = (unsigned short*)alloc((size_t)3 * 131072 * sizeof(unsigned short));

    // degrees (shared across all layers)
    zero_f32<<<1024, 256, 0, stream>>>(deg, N);
    degree_kernel<<<2048, 256, 0, stream>>>(dst, deg, E);
    invdeg_kernel<<<1024, 256, 0, stream>>>(deg, inv, N);

    // weights -> bf16 once
    for (int l = 0; l < 3; ++l)
        wconv<<<256, 256, 0, stream>>>(Wl[l], Wr[l], Wb + (size_t)l * 131072);

    const float* xin = x_all;
    float* outs[3] = {xb0, xb1, (float*)d_out};
    for (int l = 0; l < 3; ++l) {
        zero_f32<<<4096, 256, 0, stream>>>(agg, N * DFEAT);
        scatter_kernel<<<(E + 7) / 8, 256, 0, stream>>>(src, dst, xin, agg, E);
        gemm_kernel<<<(N + 127) / 128, 256, 0, stream>>>(agg, inv, xin,
                                                         Wb + (size_t)l * 131072,
                                                         bb[l], outs[l], N, (l < 2) ? 1 : 0);
        xin = outs[l];
    }
}
